// SampledSoftmax_42554535969206
// MI455X (gfx1250) — matrix operand order repaired and checked by reading, not compiled, not run on hardware
//
#include <hip/hip_runtime.h>
#include <hip/hip_bf16.h>
#include <math.h>

// ---------------------------------------------------------------------------
// Sampled-softmax loss for MI455X (gfx1250, wave32, WMMA).
//   B=32, T=512, D=1024, S=1000 sampled, C=50000 classes.
//
// Fast path (needs ~64MB scratch): one-time gather of sampled W rows into a
// bf16 buffer (L2-resident: 64MB + 64MB fp32 X < 192MB L2) + precomputed
// bias-log_q adjustments. Main GEMM streams pure-bf16 B fragments through
// v_wmma_f32_16x16x32_bf16, double-buffered B tiles, and LDS-only split
// barriers (s_wait_dscnt + s_barrier_signal/-wait) so global loads stay in
// flight across the K-partial combine; fused flash-style online logsumexp
// means the [B,T,S] logit tensor never touches memory.
// Fallback path (small scratch): fused fp32->bf16 conversion in-loop.
// ---------------------------------------------------------------------------

#define NUM_CLASSES 50000
#define NUM_SAMPLED 1000
#define D_MODEL     1024
#define BATCH       32
#define SEQ         512

typedef __attribute__((ext_vector_type(16))) __bf16 v16bf;
typedef __attribute__((ext_vector_type(8)))  __bf16 v8bf;
typedef __attribute__((ext_vector_type(8)))  float  v8f;

// 1/ln(NUM_CLASSES+1) = 1/ln(50001)
#define INV_LOG_NC1 0.0924233f
// "column disabled" additive term: logit becomes ~-1e30 -> exp() == 0
#define NEG_BIG     -1.0e30f

// Workgroup barrier that orders ONLY LDS traffic (unlike __syncthreads, which
// fences global memory and forces s_wait_loadcnt 0, draining our in-flight
// double-buffered B loads). s_wait_dscnt 0 makes this wave's DS ops visible,
// then the CDNA5 split barrier synchronizes the workgroup.
__device__ __forceinline__ void lds_barrier() {
    asm volatile("s_wait_dscnt 0x0\n\t"
                 "s_barrier_signal -1\n\t"
                 "s_barrier_wait -1" ::: "memory");
}

__device__ __forceinline__ float log_expected_count(int id) {
    // TF log-uniform sampler expected-count (unique=True), in log space.
    float idf = (float)id;
    float p = (logf(idf + 2.0f) - logf(idf + 1.0f)) * INV_LOG_NC1;
    return logf(-expm1f((float)NUM_SAMPLED * log1pf(-p)));
}

__device__ __forceinline__ void pack8(v16bf& v, int base, float4 a, float4 b) {
    v[base + 0] = (__bf16)a.x; v[base + 1] = (__bf16)a.y;
    v[base + 2] = (__bf16)a.z; v[base + 3] = (__bf16)a.w;
    v[base + 4] = (__bf16)b.x; v[base + 5] = (__bf16)b.y;
    v[base + 6] = (__bf16)b.z; v[base + 7] = (__bf16)b.w;
}

// ---------------------------------------------------------------------------
// Kernel 0 (fast path): gather sampled W rows -> bf16, precompute adjustments.
// ---------------------------------------------------------------------------
__global__ void ssm_gather_kernel(const int*   __restrict__ sampled,
                                  const float* __restrict__ W,
                                  const float* __restrict__ bias,
                                  __bf16*      __restrict__ Wg,
                                  float*       __restrict__ adj) {
    const int bs  = blockIdx.x;                 // 0 .. B*S-1
    const int sid = sampled[bs];
    const float* src = W  + (size_t)sid * D_MODEL;
    __bf16*      dst = Wg + (size_t)bs  * D_MODEL;
    const int t = threadIdx.x;
    float4 a = *(const float4*)(src + t * 8);
    float4 c = *(const float4*)(src + t * 8 + 4);
    v8bf o;
    o[0] = (__bf16)a.x; o[1] = (__bf16)a.y; o[2] = (__bf16)a.z; o[3] = (__bf16)a.w;
    o[4] = (__bf16)c.x; o[5] = (__bf16)c.y; o[6] = (__bf16)c.z; o[7] = (__bf16)c.w;
    *(v8bf*)(dst + t * 8) = o;
    if (t == 0) adj[bs] = bias[sid] - log_expected_count(sid);
}

// ---------------------------------------------------------------------------
// Kernel 1: exact fp32 true logits. One wave per (b,t) row.
// ---------------------------------------------------------------------------
__global__ void ssm_true_logit_kernel(const float* __restrict__ X,
                                      const int*   __restrict__ labels,
                                      const float* __restrict__ W,
                                      const float* __restrict__ bias,
                                      float* __restrict__ tl) {
    int row  = blockIdx.x * (blockDim.x >> 5) + (threadIdx.x >> 5);
    int lane = threadIdx.x & 31;
    if (row >= BATCH * SEQ) return;
    int lab = labels[row];
    const float* x = X + (size_t)row * D_MODEL;
    const float* w = W + (size_t)lab * D_MODEL;
    float s = 0.0f;
    #pragma unroll
    for (int d = 0; d < D_MODEL; d += 32 * 4) {
        float4 xv = *(const float4*)(x + d + lane * 4);
        float4 wv = *(const float4*)(w + d + lane * 4);
        s += xv.x * wv.x + xv.y * wv.y + xv.z * wv.z + xv.w * wv.w;
    }
    #pragma unroll
    for (int off = 16; off >= 1; off >>= 1) s += __shfl_xor(s, off, 32);
    if (lane == 0) tl[row] = s + bias[lab] - log_expected_count(lab);
}

// ---------------------------------------------------------------------------
// Helpers for the fast-path hot loop (all branch-free).
// ---------------------------------------------------------------------------
struct TileMeta { int sid; float adj; };

__device__ __forceinline__ TileMeta load_tileB(v16bf (&B)[8],
                                               const __bf16* __restrict__ Wg,
                                               const float*  __restrict__ adjbuf,
                                               const int*    __restrict__ sampled,
                                               int b, int tn, int L, int h, int k0) {
    TileMeta t;
    const int n     = tn * 16 + L;
    const bool valid = (n < NUM_SAMPLED);
    const int ncl   = valid ? n : NUM_SAMPLED - 1;      // v_cndmask, no branch
    // sid + adj issued BEFORE the B loads so no late wait drains the pipeline
    t.sid = sampled[b * NUM_SAMPLED + ncl];
    float araw = adjbuf[b * NUM_SAMPLED + ncl];         // unconditional load
    t.adj = valid ? araw : NEG_BIG;                     // select, no branch
    const __bf16* wrow = Wg + (size_t)(b * NUM_SAMPLED + ncl) * D_MODEL;
    #pragma unroll
    for (int c = 0; c < 8; ++c) {
        int kb = k0 + 32 * c + 8 * h;
        v8bf lo = *(const v8bf*)(wrow + kb);
        v8bf hi = *(const v8bf*)(wrow + kb + 16);
        B[c] = __builtin_shufflevector(lo, hi,
            0, 1, 2, 3, 4, 5, 6, 7, 8, 9, 10, 11, 12, 13, 14, 15);
    }
    return t;
}

__device__ __forceinline__ void process_tile(const v16bf (&A)[8], const v16bf (&B)[8],
                                             const TileMeta& tm, const int (&lab)[8],
                                             int wave, int lane,
                                             float (&cbuf)[4][8][32],
                                             float (&m_run)[2], float (&l_run)[2]) {
    v8f acc = {0.f, 0.f, 0.f, 0.f, 0.f, 0.f, 0.f, 0.f};
    #pragma unroll
    for (int c = 0; c < 8; ++c)
        acc = __builtin_amdgcn_wmma_f32_16x16x32_bf16(
            false, A[c], false, B[c], (short)0, acc, false, false);

    lds_barrier();                         // previous tile's cbuf reads done
    #pragma unroll
    for (int r = 0; r < 8; ++r) cbuf[wave][r][lane] = acc[r];
    lds_barrier();                         // all partials visible

    #pragma unroll
    for (int rr = 0; rr < 2; ++rr) {
        int r = 2 * wave + rr;
        float v = cbuf[0][r][lane] + cbuf[1][r][lane]
                + cbuf[2][r][lane] + cbuf[3][r][lane];
        float logit = v + tm.adj;                        // invalid -> ~-1e30
        logit = (tm.sid == lab[r]) ? -1.0e9f : logit;    // accidental hit
        float mo = m_run[rr];
        float mn = fmaxf(mo, logit);
        l_run[rr] = l_run[rr] * __expf(mo - mn) + __expf(logit - mn);
        m_run[rr] = mn;
    }
}

// ---------------------------------------------------------------------------
// Kernel 2a (fast path): bf16 WMMA + fused online logsumexp, B double-buffered.
// Block = 128 threads = 4 waves owning 16 rows of one batch; waves split K
// (256 each) so A fragments stay resident; K-partials combine via 4KB LDS.
// ---------------------------------------------------------------------------
__global__ __launch_bounds__(128)
void ssm_sampled_lse_fast(const float*  __restrict__ X,
                          const int*    __restrict__ labels,
                          const int*    __restrict__ sampled,
                          const __bf16* __restrict__ Wg,
                          const float*  __restrict__ adjbuf,
                          const float*  __restrict__ tl,
                          float* __restrict__ loss) {
    const int b    = blockIdx.y;
    const int m0   = blockIdx.x * 16;
    const int wave = threadIdx.x >> 5;
    const int lane = threadIdx.x & 31;
    const int L    = lane & 15;
    const int h    = lane >> 4;
    const int k0   = wave * (D_MODEL / 4);

    __shared__ float cbuf[4][8][32];

    // resident A fragments: 16 rows x 256 K (ISA 16-bit A layout)
    v16bf A[8];
    const float* xrow = X + ((size_t)b * SEQ + (m0 + L)) * D_MODEL;
    #pragma unroll
    for (int c = 0; c < 8; ++c) {
        int kb = k0 + 32 * c + 8 * h;
        float4 f0 = *(const float4*)(xrow + kb);
        float4 f1 = *(const float4*)(xrow + kb + 4);
        float4 f2 = *(const float4*)(xrow + kb + 16);
        float4 f3 = *(const float4*)(xrow + kb + 20);
        pack8(A[c], 0, f0, f1);
        pack8(A[c], 8, f2, f3);
    }

    int lab[8];
    #pragma unroll
    for (int r = 0; r < 8; ++r) lab[r] = labels[b * SEQ + m0 + r + 8 * h];

    float m_run[2] = { -3.0e38f, -3.0e38f };
    float l_run[2] = { 0.0f, 0.0f };

    const int NT = (NUM_SAMPLED + 15) / 16;    // 63 column tiles

    // software pipeline: B double-buffered so tile tn+1's loads fly during
    // tile tn's WMMAs + epilogue (the LDS-only barriers never drain them).
    v16bf B0[8], B1[8];
    TileMeta t0 = load_tileB(B0, Wg, adjbuf, sampled, b, 0, L, h, k0);
    TileMeta t1;
    int tn = 0;
    while (tn < NT) {
        if (tn + 1 < NT) t1 = load_tileB(B1, Wg, adjbuf, sampled, b, tn + 1, L, h, k0);
        process_tile(A, B0, t0, lab, wave, lane, cbuf, m_run, l_run);
        ++tn;
        if (tn >= NT) break;
        if (tn + 1 < NT) t0 = load_tileB(B0, Wg, adjbuf, sampled, b, tn + 1, L, h, k0);
        process_tile(A, B1, t1, lab, wave, lane, cbuf, m_run, l_run);
        ++tn;
    }

    // merge across the 16 lanes of each half, fold in the true logit
    #pragma unroll
    for (int rr = 0; rr < 2; ++rr) {
        float m = m_run[rr], l = l_run[rr];
        #pragma unroll
        for (int off = 8; off >= 1; off >>= 1) {
            float mo = __shfl_xor(m, off, 32);
            float lo = __shfl_xor(l, off, 32);
            float mn = fmaxf(m, mo);
            l = l * expf(m - mn) + lo * expf(mo - mn);
            m = mn;
        }
        if (L == 0) {
            int r   = 2 * wave + rr;
            int row = b * SEQ + m0 + r + 8 * h;
            float t  = tl[row];
            float M2 = fmaxf(t, m);
            float lse = M2 + logf(expf(t - M2) + l * expf(m - M2));
            loss[row] = lse - t;
        }
    }
}

// ---------------------------------------------------------------------------
// Kernel 2b (fallback, small scratch): fp32->bf16 conversion fused in-loop.
// ---------------------------------------------------------------------------
__global__ __launch_bounds__(128)
void ssm_sampled_lse_fused(const float*  __restrict__ X,
                           const int*    __restrict__ labels,
                           const int*    __restrict__ sampled,
                           const float*  __restrict__ W,
                           const float*  __restrict__ bias,
                           const float*  __restrict__ tl,
                           float* __restrict__ loss) {
    const int b    = blockIdx.y;
    const int m0   = blockIdx.x * 16;
    const int wave = threadIdx.x >> 5;
    const int lane = threadIdx.x & 31;
    const int L    = lane & 15;
    const int h    = lane >> 4;
    const int k0   = wave * (D_MODEL / 4);

    __shared__ float cbuf[4][8][32];

    v16bf A[8];
    const float* xrow = X + ((size_t)b * SEQ + (m0 + L)) * D_MODEL;
    #pragma unroll
    for (int c = 0; c < 8; ++c) {
        int kb = k0 + 32 * c + 8 * h;
        float4 f0 = *(const float4*)(xrow + kb);
        float4 f1 = *(const float4*)(xrow + kb + 4);
        float4 f2 = *(const float4*)(xrow + kb + 16);
        float4 f3 = *(const float4*)(xrow + kb + 20);
        pack8(A[c], 0, f0, f1);
        pack8(A[c], 8, f2, f3);
    }

    int lab[8];
    #pragma unroll
    for (int r = 0; r < 8; ++r) lab[r] = labels[b * SEQ + m0 + r + 8 * h];

    float m_run[2] = { -3.0e38f, -3.0e38f };
    float l_run[2] = { 0.0f, 0.0f };

    const int NT = (NUM_SAMPLED + 15) / 16;
    for (int tn = 0; tn < NT; ++tn) {
        const int  n     = tn * 16 + L;
        const bool valid = (n < NUM_SAMPLED);
        const int  ncl   = valid ? n : NUM_SAMPLED - 1;
        const int  sid   = sampled[b * NUM_SAMPLED + ncl];
        const float* wrow = W + (size_t)sid * D_MODEL;

        // branch-free adjustment (computed while B loads are in flight)
        float araw = bias[sid] - log_expected_count(sid);
        float adj  = valid ? araw : NEG_BIG;

        v8f acc = {0.f, 0.f, 0.f, 0.f, 0.f, 0.f, 0.f, 0.f};
        #pragma unroll
        for (int c = 0; c < 8; ++c) {
            int kb = k0 + 32 * c + 8 * h;
            float4 g0 = *(const float4*)(wrow + kb);
            float4 g1 = *(const float4*)(wrow + kb + 4);
            float4 g2 = *(const float4*)(wrow + kb + 16);
            float4 g3 = *(const float4*)(wrow + kb + 20);
            v16bf Bv;
            pack8(Bv, 0, g0, g1);
            pack8(Bv, 8, g2, g3);
            acc = __builtin_amdgcn_wmma_f32_16x16x32_bf16(
                false, A[c], false, Bv, (short)0, acc, false, false);
        }

        lds_barrier();
        #pragma unroll
        for (int r = 0; r < 8; ++r) cbuf[wave][r][lane] = acc[r];
        lds_barrier();

        #pragma unroll
        for (int rr = 0; rr < 2; ++rr) {
            int r = 2 * wave + rr;
            float v = cbuf[0][r][lane] + cbuf[1][r][lane]
                    + cbuf[2][r][lane] + cbuf[3][r][lane];
            float logit = v + adj;
            logit = (sid == lab[r]) ? -1.0e9f : logit;
            float mo = m_run[rr];
            float mn = fmaxf(mo, logit);
            l_run[rr] = l_run[rr] * __expf(mo - mn) + __expf(logit - mn);
            m_run[rr] = mn;
        }
    }

    #pragma unroll
    for (int rr = 0; rr < 2; ++rr) {
        float m = m_run[rr], l = l_run[rr];
        #pragma unroll
        for (int off = 8; off >= 1; off >>= 1) {
            float mo = __shfl_xor(m, off, 32);
            float lo = __shfl_xor(l, off, 32);
            float mn = fmaxf(m, mo);
            l = l * expf(m - mn) + lo * expf(mo - mn);
            m = mn;
        }
        if (L == 0) {
            int r   = 2 * wave + rr;
            int row = b * SEQ + m0 + r + 8 * h;
            float t  = tl[row];
            float M2 = fmaxf(t, m);
            float lse = M2 + logf(expf(t - M2) + l * expf(m - M2));
            loss[row] = lse - t;
        }
    }
}

// ---------------------------------------------------------------------------
// Kernel 3: 0.5 * mean over the 16384 per-row losses.
// ---------------------------------------------------------------------------
__global__ void ssm_reduce_kernel(const float* __restrict__ loss,
                                  float* __restrict__ out) {
    __shared__ float sm[256];
    float s = 0.0f;
    for (int i = threadIdx.x; i < BATCH * SEQ; i += 256) s += loss[i];
    sm[threadIdx.x] = s;
    __syncthreads();
    for (int st = 128; st >= 1; st >>= 1) {
        if (threadIdx.x < st) sm[threadIdx.x] += sm[threadIdx.x + st];
        __syncthreads();
    }
    if (threadIdx.x == 0) out[0] = 0.5f * sm[0] / (float)(BATCH * SEQ);
}

// ---------------------------------------------------------------------------
extern "C" void kernel_launch(void* const* d_in, const int* in_sizes, int n_in,
                              void* d_out, int out_size, void* d_ws, size_t ws_size,
                              hipStream_t stream) {
    (void)in_sizes; (void)n_in; (void)out_size;
    const float* X       = (const float*)d_in[0];   // [32,512,1024]
    const int*   labels  = (const int*)  d_in[1];   // [32,512]
    const int*   sampled = (const int*)  d_in[2];   // [32,1000]
    const float* W       = (const float*)d_in[3];   // [50000,1024]
    const float* bias    = (const float*)d_in[4];   // [50000]
    float*       out     = (float*)d_out;

    // ws layout: [tl 16384 f32][loss 16384 f32][adj 32000 f32] ... (256KB) [Wg bf16]
    float*  tl   = (float*)d_ws;
    float*  loss = tl + BATCH * SEQ;
    float*  adj  = loss + BATCH * SEQ;
    __bf16* Wg   = (__bf16*)((char*)d_ws + 256 * 1024);
    const size_t need_full = 256 * 1024 +
        (size_t)BATCH * NUM_SAMPLED * D_MODEL * sizeof(__bf16);

    // K1: exact true logits, one wave per row
    ssm_true_logit_kernel<<<(BATCH * SEQ) / 8, 256, 0, stream>>>(X, labels, W, bias, tl);

    dim3 grid(SEQ / 16, BATCH);
    if (ws_size >= need_full) {
        // K0: gather + bf16-convert sampled W rows (L2-resident thereafter)
        ssm_gather_kernel<<<BATCH * NUM_SAMPLED, 128, 0, stream>>>(sampled, W, bias, Wg, adj);
        ssm_sampled_lse_fast<<<grid, 128, 0, stream>>>(
            X, labels, sampled, Wg, adj, tl, loss);
    } else {
        ssm_sampled_lse_fused<<<grid, 128, 0, stream>>>(
            X, labels, sampled, W, bias, tl, loss);
    }

    // K3: scalar reduction
    ssm_reduce_kernel<<<1, 256, 0, stream>>>(loss, out);
}
